// StructDTI_20194936226359
// MI455X (gfx1250) — matrix operand order repaired and checked by reading, not compiled, not run on hardware
//
#include <hip/hip_runtime.h>
#include <hip/hip_bf16.h>
#include <math.h>

typedef __attribute__((ext_vector_type(16))) _Float16 v16h;
typedef __attribute__((ext_vector_type(8)))  _Float16 v8h;
typedef __attribute__((ext_vector_type(8)))  float    v8f;

#define NDN 12288
#define NPN 65536
#define EDN 49152
#define EPN 786432
#define MAXPER 256
#define MAXEPER 3072

// ---------------- layernorm (row) -> f16 padded ----------------
__global__ void layernorm_f16pad(const float* __restrict__ x, const float* __restrict__ w,
                                 const float* __restrict__ b, _Float16* __restrict__ out,
                                 int N, int F, int stride) {
  int n = blockIdx.x * blockDim.x + threadIdx.x;
  if (n >= N) return;
  const float* xr = x + (size_t)n * F;
  float mu = 0.f;
  for (int j = 0; j < F; ++j) mu += xr[j];
  mu /= (float)F;
  float var = 0.f;
  for (int j = 0; j < F; ++j) { float d = xr[j] - mu; var += d * d; }
  var /= (float)F;
  float inv = rsqrtf(var + 1e-5f);
  _Float16* o = out + (size_t)n * stride;
  for (int j = 0; j < F; ++j) o[j] = (_Float16)((xr[j] - mu) * inv * w[j] + b[j]);
  for (int j = F; j < stride; ++j) o[j] = (_Float16)0.f;
}

// ---------------- f32 [N x F] -> f16 padded [N x stride], optional elu ----------------
__global__ void f32_to_f16pad(const float* __restrict__ x, _Float16* __restrict__ out,
                              int N, int F, int stride, int do_elu) {
  int idx = blockIdx.x * blockDim.x + threadIdx.x;
  int total = N * stride;
  if (idx >= total) return;
  int n = idx / stride, j = idx - n * stride;
  float v = 0.f;
  if (j < F) {
    v = x[(size_t)n * F + j];
    if (do_elu) v = v > 0.f ? v : (expf(v) - 1.f);
  }
  out[idx] = (_Float16)v;
}

// ---------------- pack 4 [K x 64] f32 weights into B-fragment f16 layout + bias[256] ----
// B frag (16x16x32 f16): lanes 0-15 hold K=0..7,16..23 of a 16-col tile, lanes 16-31 hold K=8..15,24..31.
__global__ void pack_w(const float* __restrict__ wq, const float* __restrict__ wk,
                       const float* __restrict__ wv, const float* __restrict__ ws,
                       const float* __restrict__ bq, const float* __restrict__ bk,
                       const float* __restrict__ bv, const float* __restrict__ bs,
                       _Float16* __restrict__ Wp, float* __restrict__ bias,
                       int K, int Kpad) {
  int idx = blockIdx.x * blockDim.x + threadIdx.x;
  if (idx < 256) {
    int m = idx >> 6, c = idx & 63;
    const float* bp = m == 0 ? bq : m == 1 ? bk : m == 2 ? bv : bs;
    bias[idx] = bp[c];
  }
  int total = Kpad * 256;
  if (idx >= total) return;
  int i    = idx & 15;
  int lane = (idx >> 4) & 31;
  int tn   = idx >> 9;           // t*16 + ntile
  int t = tn >> 4, ntile = tn & 15;
  int k   = t * 32 + ((i >> 3) << 4) + ((lane >> 4) << 3) + (i & 7);
  int col = ntile * 16 + (lane & 15);
  int m = col >> 6, c = col & 63;
  const float* w = m == 0 ? wq : m == 1 ? wk : m == 2 ? wv : ws;
  float v = (k < K) ? w[(size_t)k * 64 + c] : 0.f;
  Wp[idx] = (_Float16)v;
}

// -------- WMMA GEMM: Out[N x 256] = X16 @ Wpack + bias; one wave -> 16x64 strip ---------
__global__ __launch_bounds__(32)
void wmma_gemm4(const _Float16* __restrict__ X, int strideX,
                const _Float16* __restrict__ Wp, const float* __restrict__ bias,
                float* __restrict__ Out, int Kpad) {
  int tileM = blockIdx.x;          // N/16
  int nq    = blockIdx.y;          // 0..3 -> columns nq*64 .. nq*64+63
  int lane  = threadIdx.x;
  int rowA  = tileM * 16 + (lane & 15);
  int c0    = (lane >> 4) * 8;     // ISA 16-bit A layout: two 8-half chunks per lane
  v8f acc0 = {}, acc1 = {}, acc2 = {}, acc3 = {};
  int ktiles = Kpad >> 5;
  for (int t = 0; t < ktiles; ++t) {
    const _Float16* ap = X + (size_t)rowA * strideX + t * 32 + c0;
    v8h a0 = *(const v8h*)(ap);
    v8h a1 = *(const v8h*)(ap + 16);
    v16h a;
    #pragma unroll
    for (int i = 0; i < 8; ++i) { a[i] = a0[i]; a[i + 8] = a1[i]; }
    const _Float16* bp = Wp + ((size_t)(t * 16 + nq * 4) * 32 + lane) * 16;
    v16h b0 = *(const v16h*)(bp);
    v16h b1 = *(const v16h*)(bp + 32 * 16);
    v16h b2 = *(const v16h*)(bp + 2 * 32 * 16);
    v16h b3 = *(const v16h*)(bp + 3 * 32 * 16);
    acc0 = __builtin_amdgcn_wmma_f32_16x16x32_f16(false, a, false, b0, (short)0, acc0, false, false);
    acc1 = __builtin_amdgcn_wmma_f32_16x16x32_f16(false, a, false, b1, (short)0, acc1, false, false);
    acc2 = __builtin_amdgcn_wmma_f32_16x16x32_f16(false, a, false, b2, (short)0, acc2, false, false);
    acc3 = __builtin_amdgcn_wmma_f32_16x16x32_f16(false, a, false, b3, (short)0, acc3, false, false);
  }
  int rbase = tileM * 16 + ((lane >> 4) << 3);  // C/D layout: lanes 16-31 -> M=8..15
  #pragma unroll
  for (int j = 0; j < 4; ++j) {
    v8f* accp = j == 0 ? &acc0 : j == 1 ? &acc1 : j == 2 ? &acc2 : &acc3;
    int col = (nq * 4 + j) * 16 + (lane & 15);
    float bv = bias[col];
    #pragma unroll
    for (int i = 0; i < 8; ++i)
      Out[(size_t)(rbase + i) * 256 + col] = (*accp)[i] + bv;
  }
}

// ---------------- per-graph fused attention (softmax over in-edges + scatter) -----------
__device__ inline void atomicMaxF_sh(float* addr, float val) {
  if (val >= 0.f) atomicMax((int*)addr, __float_as_int(val));
  else            atomicMin((unsigned int*)addr, __float_as_uint(val));
}

__global__ __launch_bounds__(256)
void tconv_attn_graph(const int* __restrict__ src, const int* __restrict__ dst,
                      const float* __restrict__ QK, float* __restrict__ out,
                      int per, int eper) {
  __shared__ int   cnt[MAXPER + 1];
  __shared__ int   cur[MAXPER];
  __shared__ int   csr[MAXEPER];
  __shared__ int   ssrc[MAXEPER];
  __shared__ float aval[MAXEPER * 2];
  __shared__ float nmax[MAXPER * 2];
  __shared__ float nsum[MAXPER * 2];
  int g = blockIdx.x, tid = threadIdx.x;
  int nbase = g * per;
  long ebase = (long)g * eper;
  for (int i = tid; i < per * 2; i += 256) { nmax[i] = -3.0e38f; nsum[i] = 0.f; }
  for (int i = tid; i < per + 1; i += 256) cnt[i] = 0;
  __syncthreads();
  // pass 1: logits, per-(dst,head) max, per-dst counts
  for (int e = tid; e < eper; e += 256) {
    int sN = src[ebase + e], dN = dst[ebase + e];
    int dl = dN - nbase;
    ssrc[e] = sN;
    const float* q = QK + (size_t)dN * 256;        // cols 0..63   = Q
    const float* k = QK + (size_t)sN * 256 + 64;   // cols 64..127 = K
    #pragma unroll
    for (int h = 0; h < 2; ++h) {
      float acc = 0.f;
      for (int c = 0; c < 32; ++c) acc += q[h * 32 + c] * k[h * 32 + c];
      acc *= 0.17677669529663687f;                 // 1/sqrt(32)
      aval[e * 2 + h] = acc;
      atomicMaxF_sh(&nmax[dl * 2 + h], acc);
    }
    atomicAdd(&cnt[dl + 1], 1);
  }
  __syncthreads();
  if (tid == 0) for (int i = 1; i <= per; ++i) cnt[i] += cnt[i - 1];
  __syncthreads();
  for (int i = tid; i < per; i += 256) cur[i] = cnt[i];
  __syncthreads();
  // pass 2: exp, per-(dst,head) sum, CSR placement
  for (int e = tid; e < eper; e += 256) {
    int dl = dst[ebase + e] - nbase;
    #pragma unroll
    for (int h = 0; h < 2; ++h) {
      float v = expf(aval[e * 2 + h] - nmax[dl * 2 + h]);
      aval[e * 2 + h] = v;
      atomicAdd(&nsum[dl * 2 + h], v);
    }
    int pos = atomicAdd(&cur[dl], 1);
    csr[pos] = e;
  }
  __syncthreads();
  // pass 3: atomic-free gather; 4 node slots x 64 channels
  int c = tid & 63, slot = tid >> 6, h = c >> 5;
  for (int nl = slot; nl < per; nl += 4) {
    int nG = nbase + nl;
    float acc = QK[(size_t)nG * 256 + 192 + c];    // skip = x@ws + bs (cols 192..255)
    float zinv = 1.f / (nsum[nl * 2 + h] + 1e-16f);
    int p1 = cnt[nl + 1];
    for (int p = cnt[nl]; p < p1; ++p) {
      int e = csr[p];
      acc += (aval[e * 2 + h] * zinv) * QK[(size_t)ssrc[e] * 256 + 128 + c];  // V cols 128..191
    }
    out[(size_t)nG * 64 + c] = acc;
  }
}

// ---------------- per-graph fused SAG pooling (CSR gather + score + softmax pool) -------
__global__ __launch_bounds__(256)
void sag_pool_graph(const int* __restrict__ src, const int* __restrict__ dst,
                    const float* __restrict__ x, const float* __restrict__ w_rel,
                    const float* __restrict__ w_root, const float* __restrict__ pb,
                    float* __restrict__ rep, int per, int eper) {
  __shared__ int   cnt[MAXPER + 1];
  __shared__ int   cur[MAXPER];
  __shared__ int   csr[MAXEPER];    // stores src global node id per in-edge
  __shared__ float sco[MAXPER];
  __shared__ float red[256];
  int g = blockIdx.x, tid = threadIdx.x;
  int nbase = g * per;
  long ebase = (long)g * eper;
  for (int i = tid; i < per + 1; i += 256) cnt[i] = 0;
  __syncthreads();
  for (int e = tid; e < eper; e += 256) atomicAdd(&cnt[dst[ebase + e] - nbase + 1], 1);
  __syncthreads();
  if (tid == 0) for (int i = 1; i <= per; ++i) cnt[i] += cnt[i - 1];
  __syncthreads();
  for (int i = tid; i < per; i += 256) cur[i] = cnt[i];
  __syncthreads();
  for (int e = tid; e < eper; e += 256) {
    int dl = dst[ebase + e] - nbase;
    int pos = atomicAdd(&cur[dl], 1);
    csr[pos] = src[ebase + e];
  }
  __syncthreads();
  // scores: 4 node slots x 64 channels, reduce channels per slot
  int c = tid & 63, slot = tid >> 6;
  for (int nl0 = 0; nl0 < per; nl0 += 4) {
    int nl = nl0 + slot;
    float part = 0.f;
    if (nl < per) {
      float aggc = 0.f;
      int p1 = cnt[nl + 1];
      for (int p = cnt[nl]; p < p1; ++p) aggc += x[(size_t)csr[p] * 64 + c];
      part = aggc * w_rel[c] + x[(size_t)(nbase + nl) * 64 + c] * w_root[c];
    }
    red[tid] = part; __syncthreads();
    for (int st = 32; st > 0; st >>= 1) { if (c < st) red[tid] += red[tid + st]; __syncthreads(); }
    if (c == 0 && nl < per) sco[nl] = red[slot * 64] + pb[0];
    __syncthreads();
  }
  // softmax over sco[0..per)
  float m = -3.0e38f;
  for (int i = tid; i < per; i += 256) m = fmaxf(m, sco[i]);
  red[tid] = m; __syncthreads();
  for (int st = 128; st > 0; st >>= 1) { if (tid < st) red[tid] = fmaxf(red[tid], red[tid + st]); __syncthreads(); }
  float mx = red[0]; __syncthreads();
  float z = 0.f;
  for (int i = tid; i < per; i += 256) z += expf(sco[i] - mx);
  red[tid] = z; __syncthreads();
  for (int st = 128; st > 0; st >>= 1) { if (tid < st) red[tid] += red[tid + st]; __syncthreads(); }
  float Z = red[0] + 1e-16f; __syncthreads();
  // weighted pool: slot-partial over nodes, then reduce 4 slots per channel
  float acc = 0.f;
  for (int nl = slot; nl < per; nl += 4)
    acc += (expf(sco[nl] - mx) / Z) * x[(size_t)(nbase + nl) * 64 + c];
  red[tid] = acc; __syncthreads();
  if (slot == 0) rep[(size_t)g * 64 + c] = red[c] + red[64 + c] + red[128 + c] + red[192 + c];
}

// ---------------- graph norm + elu, in place; one block/graph, thread = channel ---------
__global__ void graph_norm_elu(float* __restrict__ x, const float* __restrict__ alpha,
                               const float* __restrict__ beta, const float* __restrict__ w,
                               int per) {
  int g = blockIdx.x, c = threadIdx.x;
  int base = g * per;
  float mean = 0.f;
  for (int n = 0; n < per; ++n) mean += x[(size_t)(base + n) * 64 + c];
  mean /= (float)per;
  float am = alpha[c] * mean;
  float var = 0.f;
  for (int n = 0; n < per; ++n) { float sub = x[(size_t)(base + n) * 64 + c] - am; var += sub * sub; }
  var /= (float)per;
  float inv = rsqrtf(var + 1e-5f) * w[c];
  float bb = beta[c];
  for (int n = 0; n < per; ++n) {
    float v = (x[(size_t)(base + n) * 64 + c] - am) * inv + bb;
    x[(size_t)(base + n) * 64 + c] = v > 0.f ? v : (expf(v) - 1.f);
  }
}

// ---------------- co-attention + RESCAL + output head; one block/graph-pair -------------
__global__ void final_kernel(const float* __restrict__ rh, const float* __restrict__ rt,
                             const float* __restrict__ w_k, const float* __restrict__ w_q,
                             const float* __restrict__ cob, const float* __restrict__ coa,
                             const float* __restrict__ outw, const float* __restrict__ outb,
                             float* __restrict__ out) {
  __shared__ float H[3][64], T[3][64], keys[3][32], qrs[3][32];
  __shared__ float sc1[3][3], sc2[3][3], nH[3], nT[3], red[64];
  int g = blockIdx.x, tid = threadIdx.x;   // 64 threads
  for (int i = 0; i < 3; ++i) {
    H[i][tid] = rh[(size_t)(i * 256 + g) * 64 + tid];
    T[i][tid] = rt[(size_t)(i * 256 + g) * 64 + tid];
  }
  __syncthreads();
  if (tid < 32) {
    for (int i = 0; i < 3; ++i) {
      float a = 0.f, q = 0.f;
      for (int c = 0; c < 64; ++c) { a += H[i][c] * w_k[c * 32 + tid]; q += T[i][c] * w_q[c * 32 + tid]; }
      keys[i][tid] = a; qrs[i][tid] = q;
    }
  } else if (tid < 38) {
    int i = tid - 32;
    bool ish = i < 3; int ii = ish ? i : i - 3;
    float s = 0.f;
    for (int c = 0; c < 64; ++c) { float v = ish ? H[ii][c] : T[ii][c]; s += v * v; }
    if (ish) nH[ii] = fmaxf(sqrtf(s), 1e-12f); else nT[ii] = fmaxf(sqrtf(s), 1e-12f);
  }
  __syncthreads();
  if (tid < 9) {
    int i = tid / 3, j = tid % 3;
    float att = 0.f;
    for (int f = 0; f < 32; ++f) att += tanhf(qrs[j][f] + keys[i][f] + cob[f]) * coa[f];
    float d1 = 0.f, d2 = 0.f;
    for (int c = 0; c < 64; ++c) { d1 += H[i][c] * T[j][c]; d2 += T[i][c] * H[j][c]; }
    sc1[i][j] = att * d1 / (nH[i] * nT[j]);
    sc2[i][j] = att * d2 / (nT[i] * nH[j]);
  }
  __syncthreads();
  float f1 = 0.f, f2 = 0.f;
  for (int i = 0; i < 3; ++i)
    for (int j = 0; j < 3; ++j) { f1 += sc1[i][j] * T[j][tid]; f2 += sc2[i][j] * H[j][tid]; }
  red[tid] = fmaxf(f1, 0.f) * outw[tid] + fmaxf(f2, 0.f) * outw[64 + tid];
  __syncthreads();
  for (int st = 32; st > 0; st >>= 1) { if (tid < st) red[tid] += red[tid + st]; __syncthreads(); }
  if (tid == 0) out[g] = red[0] + outb[0];
}

// ============================== host ==============================
extern "C" void kernel_launch(void* const* d_in, const int* in_sizes, int n_in,
                              void* d_out, int out_size, void* d_ws, size_t ws_size,
                              hipStream_t stream) {
  (void)in_sizes; (void)n_in; (void)out_size; (void)ws_size;
  const float* drug_x = (const float*)d_in[0];
  const int*   dei    = (const int*)d_in[1];
  const float* prot_x = (const float*)d_in[3];
  const int*   pei    = (const int*)d_in[4];
  const int *dsrc = dei, *ddst = dei + EDN;
  const int *psrc = pei, *pdst = pei + EPN;
  // params flattened (jax pytree, dict keys sorted): blocks[3]{conv_d(bk,bq,bs,bv,wk,wq,ws,wv),
  // conv_p(same), gn(alpha,b,w), intra(same 8), pool_h(b,w_rel,w_root), pool_t(same)},
  // then co_a, co_bias, ln_d_b, ln_d_w, ln_p_b, ln_p_w, out_b, out_w, w_k, w_q
  auto F = [&](int i) { return (const float*)d_in[i]; };

  char* base = (char*)d_ws;
  size_t off = 0;
  auto alloc = [&](size_t bytes) -> void* {
    void* p = base + off; off += (bytes + 255) & ~(size_t)255; return p;
  };
  _Float16* Xd16    = (_Float16*)alloc((size_t)NDN * 96 * 2);
  _Float16* Xp16    = (_Float16*)alloc((size_t)NPN * 64 * 2);
  _Float16* Wpack   = (_Float16*)alloc((size_t)96 * 256 * 2);
  float*    bias256 = (float*)alloc(256 * 4);
  float*    QKVS    = (float*)alloc((size_t)NPN * 256 * 4);
  float*    h       = (float*)alloc((size_t)NDN * 64 * 4);
  float*    t       = (float*)alloc((size_t)NPN * 64 * 4);
  float*    rh      = (float*)alloc((size_t)3 * 256 * 64 * 4);
  float*    rt      = (float*)alloc((size_t)3 * 256 * 64 * 4);

  auto cdiv = [](long a, long b) { return (int)((a + b - 1) / b); };

  auto run_tconv = [&](const _Float16* X16, int strideX, int K, int Kpad,
                       const float* wq, const float* wk, const float* wv, const float* ws_,
                       const float* bq, const float* bk, const float* bv, const float* bs,
                       const int* src, const int* dst, int N, int per, int eper, float* hout) {
    pack_w<<<cdiv(Kpad * 256, 256), 256, 0, stream>>>(wq, wk, wv, ws_, bq, bk, bv, bs,
                                                      Wpack, bias256, K, Kpad);
    wmma_gemm4<<<dim3(N / 16, 4), 32, 0, stream>>>(X16, strideX, Wpack, bias256, QKVS, Kpad);
    tconv_attn_graph<<<256, 256, 0, stream>>>(src, dst, QKVS, hout, per, eper);
  };

  layernorm_f16pad<<<cdiv(NDN, 128), 128, 0, stream>>>(drug_x, F(108), F(107), Xd16, NDN, 70, 96);
  layernorm_f16pad<<<cdiv(NPN, 128), 128, 0, stream>>>(prot_x, F(110), F(109), Xp16, NPN, 41, 64);

  for (int b = 0; b < 3; ++b) {
    int BB = 6 + b * 33;
    // ---- drug: conv_d then intra (shared) ----
    if (b > 0)
      f32_to_f16pad<<<cdiv((long)NDN * 96, 256), 256, 0, stream>>>(h, Xd16, NDN, 64, 96, 0);
    run_tconv(Xd16, 96, b == 0 ? 70 : 64, b == 0 ? 96 : 64,
              F(BB + 5), F(BB + 4), F(BB + 7), F(BB + 6),
              F(BB + 1), F(BB + 0), F(BB + 3), F(BB + 2),
              dsrc, ddst, NDN, 48, EDN / 256, h);
    f32_to_f16pad<<<cdiv((long)NDN * 96, 256), 256, 0, stream>>>(h, Xd16, NDN, 64, 96, 1);
    run_tconv(Xd16, 96, 64, 64,
              F(BB + 24), F(BB + 23), F(BB + 26), F(BB + 25),
              F(BB + 20), F(BB + 19), F(BB + 22), F(BB + 21),
              dsrc, ddst, NDN, 48, EDN / 256, h);
    // ---- prot: conv_p then intra (shared) ----
    if (b > 0)
      f32_to_f16pad<<<cdiv((long)NPN * 64, 256), 256, 0, stream>>>(t, Xp16, NPN, 64, 64, 0);
    run_tconv(Xp16, 64, b == 0 ? 41 : 64, 64,
              F(BB + 13), F(BB + 12), F(BB + 15), F(BB + 14),
              F(BB + 9), F(BB + 8), F(BB + 11), F(BB + 10),
              psrc, pdst, NPN, 256, EPN / 256, t);
    f32_to_f16pad<<<cdiv((long)NPN * 64, 256), 256, 0, stream>>>(t, Xp16, NPN, 64, 64, 1);
    run_tconv(Xp16, 64, 64, 64,
              F(BB + 24), F(BB + 23), F(BB + 26), F(BB + 25),
              F(BB + 20), F(BB + 19), F(BB + 22), F(BB + 21),
              psrc, pdst, NPN, 256, EPN / 256, t);
    // ---- pools (before graph norm) ----
    sag_pool_graph<<<256, 256, 0, stream>>>(dsrc, ddst, h, F(BB + 28), F(BB + 29), F(BB + 27),
                                            rh + (size_t)b * 256 * 64, 48, EDN / 256);
    sag_pool_graph<<<256, 256, 0, stream>>>(psrc, pdst, t, F(BB + 31), F(BB + 32), F(BB + 30),
                                            rt + (size_t)b * 256 * 64, 256, EPN / 256);
    // ---- graph norm + elu (shared gn params) ----
    graph_norm_elu<<<256, 64, 0, stream>>>(h, F(BB + 16), F(BB + 17), F(BB + 18), 48);
    graph_norm_elu<<<256, 64, 0, stream>>>(t, F(BB + 16), F(BB + 17), F(BB + 18), 256);
  }

  final_kernel<<<256, 64, 0, stream>>>(rh, rt, F(113), F(114), F(106), F(105),
                                       F(112), F(111), (float*)d_out);
}
